// FusedMoEModularKernel_75900662054967
// MI455X (gfx1250) — compile-verified, matrix operand order checked by reading, not run on hardware
//
#include <hip/hip_runtime.h>

#define Mdim 2048
#define Kdim 2048
#define Ndim 2048
#define NE   8
#define NTOPK 2
#define MAXP (Mdim * NTOPK)   // 4096 (m, t) pairs
#define TILE_M 128
#define TILE_N 128
#define KSTEP  32
#define ASTR1  40             // GEMM1 LDS row stride (bf16 elems): 8B-aligned rows, conflict-free
#define ASTR2  48             // GEMM2 LDS row stride: 16B-aligned rows for b128 async
#define NSTEPS (Kdim / KSTEP) // 64 (Kdim == Ndim so both GEMMs use it)

typedef __attribute__((ext_vector_type(16))) __bf16 v16bf;
typedef __attribute__((ext_vector_type(8)))  float  v8f;

#define AS1 __attribute__((address_space(1)))
#define AS3 __attribute__((address_space(3)))

// GCC-style vector types matching the async-builtin pointer parameters
typedef int v2i_t __attribute__((vector_size(2 * sizeof(int))));
typedef int v4i_t __attribute__((vector_size(4 * sizeof(int))));

union Frag16 {
    v16bf bf;
    unsigned int u[8];
};

// ---- fp32 -> bf16 (round to nearest even) ---------------------------------
__device__ __forceinline__ unsigned int f2bf(float f) {
    unsigned int u = __float_as_uint(f);
    u += 0x7fffu + ((u >> 16) & 1u);
    return u >> 16;
}
__device__ __forceinline__ unsigned int pack2bf(float a, float b) {
    return f2bf(a) | (f2bf(b) << 16);
}

// ---- CDNA5 async global->LDS copies (ASYNCcnt-tracked) --------------------
__device__ __forceinline__ void async_b64(unsigned short* l, const unsigned short* g) {
#if __has_builtin(__builtin_amdgcn_global_load_async_to_lds_b64)
    __builtin_amdgcn_global_load_async_to_lds_b64(
        (AS1 v2i_t*)(unsigned long long)g,
        (AS3 v2i_t*)(unsigned)(unsigned long long)l, 0, 0);
#else
    *(uint2*)l = *(const uint2*)g;
#endif
}
__device__ __forceinline__ void async_b128(unsigned short* l, const unsigned short* g) {
#if __has_builtin(__builtin_amdgcn_global_load_async_to_lds_b128)
    __builtin_amdgcn_global_load_async_to_lds_b128(
        (AS1 v4i_t*)(unsigned long long)g,
        (AS3 v4i_t*)(unsigned)(unsigned long long)l, 0, 0);
#else
    *(uint4*)l = *(const uint4*)g;
#endif
}
template <int N>
__device__ __forceinline__ void wait_async() {
#if __has_builtin(__builtin_amdgcn_s_wait_asynccnt)
    __builtin_amdgcn_s_wait_asynccnt(N);
#else
    asm volatile("s_wait_asynccnt %0" :: "i"(N) : "memory");
#endif
}

// Copy one 16-element bf16 row segment (32B)
__device__ __forceinline__ void seg64(unsigned short* dst, const unsigned short* src) {
#pragma unroll
    for (int j = 0; j < 4; ++j) async_b64(dst + j * 4, src + j * 4);
}
__device__ __forceinline__ void seg128(unsigned short* dst, const unsigned short* src) {
    async_b128(dst, src);
    async_b128(dst + 8, src + 8);
}

// Load a 16x32 bf16 WMMA operand fragment from one LDS tile row.
// wave32 layout: lane<16 holds K={0..7,16..23}, lane>=16 holds K={8..15,24..31}.
__device__ __forceinline__ void load_frag(Frag16& f, const unsigned short* rowbase, int half) {
#pragma unroll
    for (int v = 0; v < 8; ++v) {
        const int kk = ((v & 4) << 2) + (half << 3) + ((v & 3) << 1);
        f.u[v] = *(const unsigned int*)(rowbase + kk);
    }
}

// ---------------------------------------------------------------------------
__global__ void moe_init(int* __restrict__ cnt) {
    if (threadIdx.x < NE) cnt[threadIdx.x] = 0;
}

__global__ void moe_scatter(const int* __restrict__ ids, int* __restrict__ cnt,
                            int* __restrict__ rowmap) {
    const int p = blockIdx.x * blockDim.x + threadIdx.x;
    if (p >= MAXP) return;
    const int e = ids[p];
    const int r = atomicAdd(&cnt[e], 1);
    rowmap[e * MAXP + r] = p;
}

// fp32 -> bf16 bulk conversion (8 elems / thread)
__global__ __launch_bounds__(256)
void conv_bf16(const float* __restrict__ s, unsigned short* __restrict__ d, int n8) {
    const int i = blockIdx.x * blockDim.x + threadIdx.x;
    if (i >= n8) return;
    const float4* s4 = (const float4*)s + (size_t)i * 2;
    const float4 a = s4[0], b = s4[1];
    uint4 o;
    o.x = pack2bf(a.x, a.y);
    o.y = pack2bf(a.z, a.w);
    o.z = pack2bf(b.x, b.y);
    o.w = pack2bf(b.z, b.w);
    ((uint4*)d)[i] = o;
}

// ---------------------------------------------------------------------------
// GEMM1: act[p, n] = silu(x.w1[e,n,:]) * (x.w1[e,N+n,:]); all operands bf16.
// grid (NE*32, N/128), block 256 = 8 waves (4 M-waves x 2 N-waves).
__global__ __launch_bounds__(256)
void moe_gemm1(const unsigned short* __restrict__ hbf, const unsigned short* __restrict__ w1bf,
               const int* __restrict__ cnt, const int* __restrict__ rowmap,
               unsigned short* __restrict__ act) {
    __shared__ unsigned short lA [2 * TILE_M * ASTR1];
    __shared__ unsigned short lBg[2 * TILE_N * ASTR1];
    __shared__ unsigned short lBu[2 * TILE_N * ASTR1];
    __shared__ int rowsP[TILE_M];

    const int e    = blockIdx.x >> 5;
    const int tile = blockIdx.x & 31;
    const int nc   = cnt[e];
    const int r0   = tile * TILE_M;
    if (r0 >= nc) return;
    const int n0 = blockIdx.y * TILE_N;

    const int tid  = threadIdx.x;
    const int lane = tid & 31;
    const int wave = tid >> 5;
    const int half = lane >> 4;
    const int l16  = lane & 15;
    const int waveRow = (wave >> 1) * 32;
    const int waveCol = (wave & 1) * 64;

    if (tid < TILE_M) {
        const int r = r0 + tid;
        rowsP[tid] = (r < nc) ? rowmap[e * MAXP + r] : -1;
    }
    __syncthreads();

    const int lrow  = tid >> 1;
    const int cbase = (tid & 1) * 16;
    const int pA = rowsP[lrow];
    const int mA = (pA >= 0) ? (pA >> 1) : 0;
    const unsigned short* srcA = hbf + (size_t)mA * Kdim + cbase;
    const unsigned short* srcG = w1bf + ((size_t)e * 2 * Ndim + (size_t)(n0 + lrow)) * Kdim + cbase;
    const unsigned short* srcU = w1bf + ((size_t)e * 2 * Ndim + (size_t)(Ndim + n0 + lrow)) * Kdim + cbase;
    const int doff = lrow * ASTR1 + cbase;
    const int BUF = TILE_M * ASTR1;

    const v8f vzero = {0.f, 0.f, 0.f, 0.f, 0.f, 0.f, 0.f, 0.f};
    v8f accG[2][4], accU[2][4];
#pragma unroll
    for (int mt = 0; mt < 2; ++mt)
#pragma unroll
        for (int nt = 0; nt < 4; ++nt) { accG[mt][nt] = vzero; accU[mt][nt] = vzero; }

    // prologue: async-stage step 0 into buffer 0
    seg64(lA + doff, srcA);
    seg64(lBg + doff, srcG);
    seg64(lBu + doff, srcU);

    for (int s = 0; s < NSTEPS; ++s) {
        const int cur = (s & 1) * BUF;
        if (s + 1 < NSTEPS) {
            const int nxt = ((s + 1) & 1) * BUF;
            const int k0 = (s + 1) * KSTEP;
            seg64(lA + nxt + doff, srcA + k0);
            seg64(lBg + nxt + doff, srcG + k0);
            seg64(lBu + nxt + doff, srcU + k0);
            wait_async<12>();   // current tile (12 older asyncs) complete; next stays in flight
        } else {
            wait_async<0>();
        }
        __syncthreads();

        Frag16 aF[2], bG[4], bU[4];
#pragma unroll
        for (int mt = 0; mt < 2; ++mt)
            load_frag(aF[mt], lA + cur + (waveRow + mt * 16 + l16) * ASTR1, half);
#pragma unroll
        for (int nt = 0; nt < 4; ++nt) {
            load_frag(bG[nt], lBg + cur + (waveCol + nt * 16 + l16) * ASTR1, half);
            load_frag(bU[nt], lBu + cur + (waveCol + nt * 16 + l16) * ASTR1, half);
        }
#pragma unroll
        for (int mt = 0; mt < 2; ++mt)
#pragma unroll
            for (int nt = 0; nt < 4; ++nt) {
                accG[mt][nt] = __builtin_amdgcn_wmma_f32_16x16x32_bf16(
                    false, aF[mt].bf, false, bG[nt].bf, (short)0, accG[mt][nt], false, false);
                accU[mt][nt] = __builtin_amdgcn_wmma_f32_16x16x32_bf16(
                    false, aF[mt].bf, false, bU[nt].bf, (short)0, accU[mt][nt], false, false);
            }
        __syncthreads();
    }

    // Epilogue: fast silu(gate)*up (v_exp + v_rcp), scatter bf16 by pair id.
#pragma unroll
    for (int mt = 0; mt < 2; ++mt)
#pragma unroll
        for (int nt = 0; nt < 4; ++nt) {
            const v8f g = accG[mt][nt];
            const v8f u = accU[mt][nt];
            const int ncol = n0 + waveCol + nt * 16 + l16;
#pragma unroll
            for (int i = 0; i < 8; ++i) {
                const int ml = waveRow + mt * 16 + i + 8 * half;
                const int p  = rowsP[ml];
                if (p >= 0) {
                    const float gv  = g[i];
                    const float sig = __builtin_amdgcn_rcpf(1.f + __expf(-gv));
                    act[(size_t)p * Ndim + ncol] = (unsigned short)f2bf(gv * sig * u[i]);
                }
            }
        }
}

// ---------------------------------------------------------------------------
// GEMM2: h2[p, k] = act[p,:] . w2[e,k,:]; b128 async staging.
__global__ __launch_bounds__(256)
void moe_gemm2(const unsigned short* __restrict__ act, const unsigned short* __restrict__ w2bf,
               const int* __restrict__ cnt, const int* __restrict__ rowmap,
               float* __restrict__ h2) {
    __shared__ unsigned short lA[2 * TILE_M * ASTR2];
    __shared__ unsigned short lB[2 * TILE_N * ASTR2];
    __shared__ int rowsP[TILE_M];

    const int e    = blockIdx.x >> 5;
    const int tile = blockIdx.x & 31;
    const int nc   = cnt[e];
    const int r0   = tile * TILE_M;
    if (r0 >= nc) return;
    const int k0out = blockIdx.y * TILE_N;

    const int tid  = threadIdx.x;
    const int lane = tid & 31;
    const int wave = tid >> 5;
    const int half = lane >> 4;
    const int l16  = lane & 15;
    const int waveRow = (wave >> 1) * 32;
    const int waveCol = (wave & 1) * 64;

    if (tid < TILE_M) {
        const int r = r0 + tid;
        rowsP[tid] = (r < nc) ? rowmap[e * MAXP + r] : -1;
    }
    __syncthreads();

    const int lrow  = tid >> 1;
    const int cbase = (tid & 1) * 16;
    const int pA = rowsP[lrow];
    const int pAct = (pA >= 0) ? pA : 0;
    const unsigned short* srcA = act + (size_t)pAct * Ndim + cbase;
    const unsigned short* srcB = w2bf + ((size_t)e * Kdim + (size_t)(k0out + lrow)) * Ndim + cbase;
    const int doff = lrow * ASTR2 + cbase;
    const int BUF = TILE_M * ASTR2;

    const v8f vzero = {0.f, 0.f, 0.f, 0.f, 0.f, 0.f, 0.f, 0.f};
    v8f acc[2][4];
#pragma unroll
    for (int mt = 0; mt < 2; ++mt)
#pragma unroll
        for (int nt = 0; nt < 4; ++nt) acc[mt][nt] = vzero;

    seg128(lA + doff, srcA);
    seg128(lB + doff, srcB);

    for (int s = 0; s < NSTEPS; ++s) {
        const int cur = (s & 1) * BUF;
        if (s + 1 < NSTEPS) {
            const int nxt = ((s + 1) & 1) * BUF;
            const int nn = (s + 1) * KSTEP;
            seg128(lA + nxt + doff, srcA + nn);
            seg128(lB + nxt + doff, srcB + nn);
            wait_async<4>();
        } else {
            wait_async<0>();
        }
        __syncthreads();

        Frag16 aF[2], bF[4];
#pragma unroll
        for (int mt = 0; mt < 2; ++mt)
            load_frag(aF[mt], lA + cur + (waveRow + mt * 16 + l16) * ASTR2, half);
#pragma unroll
        for (int nt = 0; nt < 4; ++nt)
            load_frag(bF[nt], lB + cur + (waveCol + nt * 16 + l16) * ASTR2, half);
#pragma unroll
        for (int mt = 0; mt < 2; ++mt)
#pragma unroll
            for (int nt = 0; nt < 4; ++nt)
                acc[mt][nt] = __builtin_amdgcn_wmma_f32_16x16x32_bf16(
                    false, aF[mt].bf, false, bF[nt].bf, (short)0, acc[mt][nt], false, false);
        __syncthreads();
    }

#pragma unroll
    for (int mt = 0; mt < 2; ++mt)
#pragma unroll
        for (int nt = 0; nt < 4; ++nt) {
            const v8f c = acc[mt][nt];
            const int kcol = k0out + waveCol + nt * 16 + l16;
#pragma unroll
            for (int i = 0; i < 8; ++i) {
                const int ml = waveRow + mt * 16 + i + 8 * half;
                const int p  = rowsP[ml];
                if (p >= 0) h2[(size_t)p * Kdim + kcol] = c[i];
            }
        }
}

// ---------------------------------------------------------------------------
__global__ __launch_bounds__(256)
void moe_combine(const float* __restrict__ h2, const float* __restrict__ tw,
                 float* __restrict__ out) {
    const int idx = blockIdx.x * blockDim.x + threadIdx.x;   // one float4 each
    const int m  = idx / (Kdim / 4);
    const int k4 = idx % (Kdim / 4);
    const float w0 = tw[m * 2 + 0];
    const float w1 = tw[m * 2 + 1];
    const float4 a = ((const float4*)(h2 + (size_t)(2 * m)     * Kdim))[k4];
    const float4 b = ((const float4*)(h2 + (size_t)(2 * m + 1) * Kdim))[k4];
    float4 o;
    o.x = w0 * a.x + w1 * b.x;
    o.y = w0 * a.y + w1 * b.y;
    o.z = w0 * a.z + w1 * b.z;
    o.w = w0 * a.w + w1 * b.w;
    ((float4*)(out + (size_t)m * Kdim))[k4] = o;
}

// ---------------------------------------------------------------------------
extern "C" void kernel_launch(void* const* d_in, const int* in_sizes, int n_in,
                              void* d_out, int out_size, void* d_ws, size_t ws_size,
                              hipStream_t stream) {
    const float* hs = (const float*)d_in[0];   // (M, K)
    const float* w1 = (const float*)d_in[1];   // (E, 2N, K)
    const float* w2 = (const float*)d_in[2];   // (E, K, N)
    const float* tw = (const float*)d_in[3];   // (M, TOPK)
    const int*   ids = (const int*)d_in[4];    // (M, TOPK)
    float* out = (float*)d_out;

    char* wsb = (char*)d_ws;
    int* cnt    = (int*)wsb;                                       // E ints
    int* rowmap = (int*)(wsb + 256);                               // E*MAXP ints
    unsigned short* hbf  = (unsigned short*)(wsb + 131328);        // M*K bf16      (8 MB)
    unsigned short* w1bf = (unsigned short*)(wsb + 8519936);       // E*2N*K bf16 (128 MB)
    unsigned short* w2bf = (unsigned short*)(wsb + 142737664);     // E*K*N bf16   (64 MB)
    unsigned short* act  = (unsigned short*)(wsb + 209846528);     // MAXP*N bf16  (16 MB)
    float* h2 = (float*)(wsb + 226623744);                         // MAXP*K f32   (32 MB)

    moe_init   <<<1, 64, 0, stream>>>(cnt);
    moe_scatter<<<MAXP / 256, 256, 0, stream>>>(ids, cnt, rowmap);
    conv_bf16  <<<(Mdim * Kdim / 8) / 256, 256, 0, stream>>>(hs, hbf, Mdim * Kdim / 8);
    conv_bf16  <<<(NE * 2 * Ndim * Kdim / 8) / 256, 256, 0, stream>>>(w1, w1bf, NE * 2 * Ndim * Kdim / 8);
    conv_bf16  <<<(NE * Kdim * Ndim / 8) / 256, 256, 0, stream>>>(w2, w2bf, NE * Kdim * Ndim / 8);
    moe_gemm1  <<<dim3(NE * (MAXP / TILE_M), Ndim / TILE_N), 256, 0, stream>>>(hbf, w1bf, cnt, rowmap, act);
    moe_gemm2  <<<dim3(NE * (MAXP / TILE_M), Kdim / TILE_N), 256, 0, stream>>>(act, w2bf, cnt, rowmap, h2);
    moe_combine<<<(Mdim * (Kdim / 4)) / 256, 256, 0, stream>>>(h2, tw, out);
}
